// T2GFormer_45638322487492
// MI455X (gfx1250) — matrix operand order repaired and checked by reading, not compiled, not run on hardware
//
#include <hip/hip_runtime.h>
#include <math.h>

// ---------------------------------------------------------------------------
// Problem constants (match reference)
// ---------------------------------------------------------------------------
#define BATCH   1024
#define NCOLS   128
#define DMODEL  512
#define NHEAD   8
#define DHEAD   64
#define DCOL    14
#define MTOK    (BATCH * NCOLS)                     // 131072 token rows
#define OUT0    ((size_t)BATCH * NCOLS * DMODEL)    // floats in "out", then fr_graph

typedef __bf16 bf16;
typedef __attribute__((ext_vector_type(16))) __bf16 bf16x16;
typedef __attribute__((ext_vector_type(8)))  __bf16 bf16x8;
typedef __attribute__((ext_vector_type(8)))  float  f32x8;
typedef __attribute__((ext_vector_type(4)))  int    i32x4;

union ABFrag { bf16x16 v; bf16x8 h[2]; };

__device__ __forceinline__ f32x8 wmma_bf16(const ABFrag& a, const ABFrag& b, f32x8 c) {
    // (neg_a, A, neg_b, B, c_mod, C, reuse_a, reuse_b)
    return __builtin_amdgcn_wmma_f32_16x16x32_bf16(false, a.v, false, b.v,
                                                   (short)0, c, false, false);
}

// ---------------------------------------------------------------------------
// CDNA5 async global->LDS copy (ASYNCcnt-tracked), with sync fallback.
// Builtin signature (from hipcc diagnostic): first param is AS1 int4*.
// ---------------------------------------------------------------------------
#if defined(__gfx1250__) && __has_builtin(__builtin_amdgcn_global_load_async_to_lds_b128)
#define GEMM_ASYNC 1
#else
#define GEMM_ASYNC 0
#endif

#define AS1 __attribute__((address_space(1)))
#define AS3 __attribute__((address_space(3)))

__device__ __forceinline__ void copy16_g2l(const bf16* g, bf16* l) {
#if GEMM_ASYNC
    __builtin_amdgcn_global_load_async_to_lds_b128(
        (AS1 i32x4*)(AS1 void*)g,
        (AS3 i32x4*)(AS3 void*)l,
        0, 0);
#else
    *(bf16x8*)l = *(const bf16x8*)g;
#endif
}

__device__ __forceinline__ void wait_async_all() {
#if GEMM_ASYNC
#if __has_builtin(__builtin_amdgcn_s_wait_asynccnt)
    __builtin_amdgcn_s_wait_asynccnt(0);
#else
    asm volatile("s_wait_asynccnt 0" ::: "memory");
#endif
#endif
}

// ---------------------------------------------------------------------------
// f32 -> bf16 conversion (x is read exactly once -> non-temporal loads)
// ---------------------------------------------------------------------------
__global__ void cvt_bf16_kernel(const float* __restrict__ src, bf16* __restrict__ dst, long n) {
    long i = (long)blockIdx.x * blockDim.x + threadIdx.x;
    if (i < n) dst[i] = (bf16)__builtin_nontemporal_load(&src[i]);
}

// ---------------------------------------------------------------------------
// Adjacency mask: mask[h,n,m] = adj ? 0 : -10000
// adj = (n!=m) && (m!=0) && (cos(col_head[h,n], col_tail[h,m]) + bias > 0)
// (sigmoid(t)>0.5 <=> t>0; forward value of STE == hard threshold)
// ---------------------------------------------------------------------------
__global__ void adj_kernel(const float* __restrict__ col_head,
                           const float* __restrict__ col_tail,
                           const float* __restrict__ bias,
                           float* __restrict__ adjm) {
    int idx = blockIdx.x * blockDim.x + threadIdx.x;
    if (idx >= NHEAD * NCOLS * NCOLS) return;
    int m = idx & (NCOLS - 1);
    int n = (idx >> 7) & (NCOLS - 1);
    int h = idx >> 14;
    const float* a = col_head + ((size_t)h * NCOLS + n) * DCOL;
    const float* b = col_tail + ((size_t)h * NCOLS + m) * DCOL;
    float dot = 0.f, na = 0.f, nb = 0.f;
#pragma unroll
    for (int d = 0; d < DCOL; ++d) {
        dot += a[d] * b[d];
        na  += a[d] * a[d];
        nb  += b[d] * b[d];
    }
    float denom = fmaxf(sqrtf(na), 1e-12f) * fmaxf(sqrtf(nb), 1e-12f);
    float t = dot / denom + bias[0];
    bool adj = (n != m) && (m != 0) && (t > 0.f);
    adjm[idx] = adj ? 0.f : -10000.f;
}

// ---------------------------------------------------------------------------
// Blocked bf16 WMMA GEMM: C[M,N] = A[M,K] @ W[N,K]^T + bias[N]
// Block tile 128(M) x 256(N), 8 waves in 2(M) x 4(N), wave tile 64x64
// (4x4 WMMA tiles), K-step 32. Operands double-buffered in LDS via async
// global->LDS copies (rows padded to 40 bf16 for conflict-free ds_load_b128).
// Epilogue modes:
//   0: store ft (raw) + fh_scaled (* rel_emb[h,d] * 0.125) in [b,h,n,d] bf16
//   1: store fv transposed in [b,h,d,n] bf16
//   2: store f32 to final output (row*N + col), non-temporal
// ---------------------------------------------------------------------------
#define LDSPAD 40   // 32 payload + 8 pad bf16 per row (80 B)

__device__ __forceinline__ void issue_stage(const bf16* __restrict__ A,
                                            const bf16* __restrict__ W,
                                            int K, int m_blk, int n_blk, int kk,
                                            bf16 (*Al)[LDSPAD], bf16 (*Wl)[LDSPAD],
                                            int t) {
    // A block: 128 rows x 32 k = 512 16-byte chunks; 2 per thread
#pragma unroll
    for (int i = 0; i < 2; ++i) {
        int c = t + i * 256;
        int row = c >> 2, cc = (c & 3) * 8;
        copy16_g2l(A + (size_t)(m_blk + row) * K + kk + cc, &Al[row][cc]);
    }
    // W block: 256 rows x 32 k = 1024 chunks; 4 per thread
#pragma unroll
    for (int i = 0; i < 4; ++i) {
        int c = t + i * 256;
        int row = c >> 2, cc = (c & 3) * 8;
        copy16_g2l(W + (size_t)(n_blk + row) * K + kk + cc, &Wl[row][cc]);
    }
}

__global__ __launch_bounds__(256)
void gemm_bf16_kernel(const bf16* __restrict__ A, const bf16* __restrict__ W,
                      const float* __restrict__ bias, const float* __restrict__ rel,
                      bf16* __restrict__ o0, bf16* __restrict__ o1,
                      float* __restrict__ o2,
                      int M, int N, int K, int mode) {
    __shared__ __align__(16) bf16 Al[2][128][LDSPAD];
    __shared__ __align__(16) bf16 Wl[2][256][LDSPAD];

    const int t    = threadIdx.x;
    const int lane = t & 31;
    const int w    = t >> 5;
    const int r    = lane & 15;
    const int hi   = lane >> 4;
    const int m_blk = blockIdx.x * 128;
    const int n_blk = blockIdx.y * 256;
    const int m0 = m_blk + (w & 1) * 64;    // wave 64x64 tile origin
    const int n0 = n_blk + (w >> 1) * 64;

    f32x8 acc[4][4] = {};

    issue_stage(A, W, K, m_blk, n_blk, 0, Al[0], Wl[0], t);

    const int NT = K >> 5;
    for (int kt = 0; kt < NT; ++kt) {
        wait_async_all();      // my async copies for stage kt complete
        __syncthreads();       // => everyone's copies done; prior compute done

        if (kt + 1 < NT)
            issue_stage(A, W, K, m_blk, n_blk, (kt + 1) * 32,
                        Al[(kt + 1) & 1], Wl[(kt + 1) & 1], t);

        const int buf = kt & 1;
        ABFrag a[4];
#pragma unroll
        for (int mi = 0; mi < 4; ++mi) {
            const bf16* p = &Al[buf][(m0 - m_blk) + mi * 16 + r][hi * 8];
            a[mi].h[0] = *(const bf16x8*)p;
            a[mi].h[1] = *(const bf16x8*)(p + 16);
        }
#pragma unroll
        for (int ni = 0; ni < 4; ++ni) {
            const bf16* q = &Wl[buf][(n0 - n_blk) + ni * 16 + r][hi * 8];
            ABFrag b;
            b.h[0] = *(const bf16x8*)q;
            b.h[1] = *(const bf16x8*)(q + 16);
#pragma unroll
            for (int mi = 0; mi < 4; ++mi)
                acc[mi][ni] = wmma_bf16(a[mi], b, acc[mi][ni]);
        }
    }

    // ---- epilogue ---------------------------------------------------------
#pragma unroll
    for (int mi = 0; mi < 4; ++mi) {
#pragma unroll
        for (int ni = 0; ni < 4; ++ni) {
#pragma unroll
            for (int j = 0; j < 8; ++j) {
                int row = m0 + mi * 16 + j + hi * 8;    // token row
                int col = n0 + ni * 16 + r;             // feature col
                float v = acc[mi][ni][j] + bias[col];
                if (mode == 2) {
                    __builtin_nontemporal_store(v, &o2[(size_t)row * N + col]);
                } else {
                    int bidx = row >> 7, n = row & (NCOLS - 1);
                    int h = col >> 6, d = col & (DHEAD - 1);
                    if (mode == 0) {
                        size_t idx = (((size_t)bidx * NHEAD + h) * NCOLS + n) * DHEAD + d;
                        o0[idx] = (bf16)v;                                    // ft (raw)
                        o1[idx] = (bf16)(v * rel[h * DHEAD + d] * 0.125f);    // fh*rel/sqrt(64)
                    } else {                                                  // mode 1: fv^T
                        size_t idx = (((size_t)bidx * NHEAD + h) * DHEAD + d) * NCOLS + n;
                        o0[idx] = (bf16)v;
                    }
                }
            }
        }
    }
}

// ---------------------------------------------------------------------------
// Attention: one block per (b,h); 8 waves, wave w owns rows 16w..16w+15.
// S = fh_scaled @ ft^T (+mask), row softmax -> fr_graph (f32, non-temporal)
// and P (bf16 via LDS register-layout transpose), then O = P @ fv -> ctx.
// ---------------------------------------------------------------------------
__global__ __launch_bounds__(256)
void attn_kernel(const bf16* __restrict__ fhs, const bf16* __restrict__ ftb,
                 const bf16* __restrict__ fvT, const float* __restrict__ adjm,
                 float* __restrict__ fr_out, bf16* __restrict__ ctx) {
    __shared__ __align__(16) bf16 Plds[8][16][NCOLS];

    const int lane = threadIdx.x & 31;
    const int w    = threadIdx.x >> 5;
    const int r    = lane & 15;
    const int hi   = lane >> 4;
    const int bh   = blockIdx.x;          // b*NHEAD + h
    const int b    = bh >> 3;
    const int h    = bh & (NHEAD - 1);

    const bf16* Afh = fhs + (size_t)bh * NCOLS * DHEAD;
    const bf16* Bft = ftb + (size_t)bh * NCOLS * DHEAD;

    // ---- S = fh_scaled @ ft^T : wave rows 16w..16w+15, all 128 cols -------
    f32x8 s[8] = {};
#pragma unroll
    for (int kt = 0; kt < 2; ++kt) {
        const int k0 = kt * 32;
        ABFrag a;
        const bf16* p = Afh + (size_t)(w * 16 + r) * DHEAD + k0 + hi * 8;
        a.h[0] = *(const bf16x8*)p;
        a.h[1] = *(const bf16x8*)(p + 16);
#pragma unroll
        for (int nt = 0; nt < 8; ++nt) {
            const bf16* q = Bft + (size_t)(nt * 16 + r) * DHEAD + k0 + hi * 8;
            ABFrag bb;
            bb.h[0] = *(const bf16x8*)q;
            bb.h[1] = *(const bf16x8*)(q + 16);
            s[nt] = wmma_bf16(a, bb, s[nt]);
        }
    }

    // ---- mask + softmax per row (row split: 16 lanes x 8 col-tiles) -------
    const float* am = adjm + (size_t)h * NCOLS * NCOLS;
    float* fr = fr_out + (size_t)bh * NCOLS * NCOLS;
#pragma unroll
    for (int j = 0; j < 8; ++j) {
        const int row = w * 16 + j + hi * 8;
        float v[8];
        float mx = -3.0e38f;
#pragma unroll
        for (int nt = 0; nt < 8; ++nt) {
            const int col = nt * 16 + r;
            v[nt] = s[nt][j] + am[row * NCOLS + col];
            mx = fmaxf(mx, v[nt]);
        }
        for (int off = 1; off < 16; off <<= 1) mx = fmaxf(mx, __shfl_xor(mx, off, 32));
        float sum = 0.f;
#pragma unroll
        for (int nt = 0; nt < 8; ++nt) { v[nt] = __expf(v[nt] - mx); sum += v[nt]; }
        for (int off = 1; off < 16; off <<= 1) sum += __shfl_xor(sum, off, 32);
        const float inv = 1.0f / sum;
#pragma unroll
        for (int nt = 0; nt < 8; ++nt) {
            const float p = v[nt] * inv;
            const int col = nt * 16 + r;
            __builtin_nontemporal_store(p, &fr[(size_t)row * NCOLS + col]);  // fr_graph
            Plds[w][j + hi * 8][col] = (bf16)p;         // C-layout -> A-layout via LDS
        }
    }
    __syncthreads();

    // ---- O = P @ fv : K=128 over tokens, 64 d-cols ------------------------
    const bf16* Bfv = fvT + (size_t)bh * DHEAD * NCOLS;   // [d, token] per head
    f32x8 o[4] = {};
#pragma unroll
    for (int kt = 0; kt < 4; ++kt) {
        const int k0 = kt * 32;
        ABFrag a;
        const bf16* p = &Plds[w][r][0] + k0 + hi * 8;
        a.h[0] = *(const bf16x8*)p;
        a.h[1] = *(const bf16x8*)(p + 16);
#pragma unroll
        for (int dt = 0; dt < 4; ++dt) {
            const bf16* q = Bfv + (size_t)(dt * 16 + r) * NCOLS + k0 + hi * 8;
            ABFrag bb;
            bb.h[0] = *(const bf16x8*)q;
            bb.h[1] = *(const bf16x8*)(q + 16);
            o[dt] = wmma_bf16(a, bb, o[dt]);
        }
    }

    bf16* C = ctx + (size_t)b * NCOLS * DMODEL;
#pragma unroll
    for (int dt = 0; dt < 4; ++dt) {
#pragma unroll
        for (int j = 0; j < 8; ++j) {
            const int row = w * 16 + j + hi * 8;          // token in column block
            const int d   = dt * 16 + r;
            C[(size_t)row * DMODEL + h * DHEAD + d] = (bf16)o[dt][j];
        }
    }
}

// ---------------------------------------------------------------------------
// Host-side launch
// ---------------------------------------------------------------------------
extern "C" void kernel_launch(void* const* d_in, const int* in_sizes, int n_in,
                              void* d_out, int out_size, void* d_ws, size_t ws_size,
                              hipStream_t stream) {
    (void)in_sizes; (void)n_in; (void)out_size; (void)ws_size;

    const float* x        = (const float*)d_in[0];
    const float* W_head_w = (const float*)d_in[1];
    const float* W_head_b = (const float*)d_in[2];
    const float* W_v_w    = (const float*)d_in[3];
    const float* W_v_b    = (const float*)d_in[4];
    const float* W_out_w  = (const float*)d_in[5];
    const float* W_out_b  = (const float*)d_in[6];
    const float* rel_emb  = (const float*)d_in[7];
    const float* col_head = (const float*)d_in[8];
    const float* col_tail = (const float*)d_in[9];
    const float* bias     = (const float*)d_in[10];

    float* outF = (float*)d_out;            // [131072, 512]
    float* frF  = outF + OUT0;              // [1024, 8, 128, 128]

    // Workspace carve-up
    const long XN = (long)MTOK * DMODEL;    // 67108864
    const long WN = (long)DMODEL * DMODEL;  // 262144
    char* ws = (char*)d_ws;
    bf16* xb  = (bf16*)ws;                      ws += XN * 2;
    bf16* wh  = (bf16*)ws;                      ws += WN * 2;
    bf16* wv  = (bf16*)ws;                      ws += WN * 2;
    bf16* wo  = (bf16*)ws;                      ws += WN * 2;
    bf16* fhs = (bf16*)ws;                      ws += XN * 2;  // fh * rel / 8   [b,h,n,d]
    bf16* ftb = (bf16*)ws;                      ws += XN * 2;  // f_head raw     [b,h,n,d]
    bf16* fvT = (bf16*)ws;                      ws += XN * 2;  // f_v transposed [b,h,d,n]
    bf16* ctx = (bf16*)ws;                      ws += XN * 2;  // attention out  [b,n,512]
    float* adjm = (float*)ws;                                  // [8,128,128]

    // 1) precision conversion
    cvt_bf16_kernel<<<(unsigned)((XN + 255) / 256), 256, 0, stream>>>(x, xb, XN);
    cvt_bf16_kernel<<<(unsigned)((WN + 255) / 256), 256, 0, stream>>>(W_head_w, wh, WN);
    cvt_bf16_kernel<<<(unsigned)((WN + 255) / 256), 256, 0, stream>>>(W_v_w, wv, WN);
    cvt_bf16_kernel<<<(unsigned)((WN + 255) / 256), 256, 0, stream>>>(W_out_w, wo, WN);

    // 2) learned topology -> additive mask
    adj_kernel<<<(NHEAD * NCOLS * NCOLS) / 256, 256, 0, stream>>>(col_head, col_tail, bias, adjm);

    // 3) projections (WMMA GEMMs, LDS double-buffered, async staged)
    dim3 ggrid(MTOK / 128, DMODEL / 256);
    gemm_bf16_kernel<<<ggrid, 256, 0, stream>>>(xb, wh, W_head_b, rel_emb,
                                                ftb, fhs, nullptr,
                                                MTOK, DMODEL, DMODEL, /*mode=*/0);
    gemm_bf16_kernel<<<ggrid, 256, 0, stream>>>(xb, wv, W_v_b, rel_emb,
                                                fvT, nullptr, nullptr,
                                                MTOK, DMODEL, DMODEL, /*mode=*/1);

    // 4) attention (scores + softmax + fr_graph + context), one block per (b,h)
    attn_kernel<<<BATCH * NHEAD, 256, 0, stream>>>(fhs, ftb, fvT, adjm, frF, ctx);

    // 5) output projection -> final f32 output
    gemm_bf16_kernel<<<ggrid, 256, 0, stream>>>(ctx, wo, W_out_b, rel_emb,
                                                nullptr, nullptr, outF,
                                                MTOK, DMODEL, DMODEL, /*mode=*/2);
}
